// NanoShakes_43885975830729
// MI455X (gfx1250) — compile-verified, matrix operand order checked by reading, not run on hardware
//
#include <hip/hip_runtime.h>

typedef unsigned short u16;
typedef __attribute__((ext_vector_type(16))) __bf16 v16bf;
typedef __attribute__((ext_vector_type(8)))  float  v8f;

#define FLAG_RELU  1
#define FLAG_RESID 2
#define FLAG_BF16  4
#define FLAG_BIAS  8

__device__ __forceinline__ u16 f32_to_bf16(float f) {
  unsigned int u = __float_as_uint(f);
  u += 0x7FFFu + ((u >> 16) & 1u);   // round-to-nearest-even
  return (u16)(u >> 16);
}

// Build one 16x32 bf16 WMMA operand fragment from a row-major [rows][stride]
// bf16 buffer. Per ISA 7.12.2: lane<16 -> K {0..7,16..23}, lane>=16 -> K {8..15,24..31}.
__device__ __forceinline__ v16bf ld_frag(const u16* p0, int stride, int row, int kbase, int lane) {
  const u16* p = p0 + (size_t)row * stride + kbase + ((lane & 16) ? 8 : 0);
  union { v16bf v; uint4 u[2]; } f;
  f.u[0] = *reinterpret_cast<const uint4*>(p);
  f.u[1] = *reinterpret_cast<const uint4*>(p + 16);
  return f.v;
}

#define WMMA_BF16(a, b, c) \
  __builtin_amdgcn_wmma_f32_16x16x32_bf16(false, (a), false, (b), (short)0, (c), false, false)

// CDNA5 async global->LDS copy of 16 bytes per lane (GLOBAL_LOAD_ASYNC_TO_LDS_B128,
// tracked with ASYNCcnt). vdst = LDS byte offset in a VGPR, vaddr = 64-bit global addr.
__device__ __forceinline__ void async_ld16(unsigned lds_off, const u16* g) {
  asm volatile("global_load_async_to_lds_b128 %0, %1, off"
               :: "v"(lds_off), "v"((unsigned long long)(size_t)g)
               : "memory");
}
#define WAIT_ASYNC(n) asm volatile("s_wait_asynccnt " #n ::: "memory")

// ---------------------------------------------------------------------------
// Weight transpose + fp32->bf16 convert: in[K][N] (fp32) -> out[N][K] (bf16)
// ---------------------------------------------------------------------------
__global__ __launch_bounds__(256) void transpose_convert_kernel(
    const float* __restrict__ in, u16* __restrict__ out, int K, int N) {
  __shared__ float tile[32][33];
  int n0 = blockIdx.x * 32, k0 = blockIdx.y * 32;
  int tx = threadIdx.x, ty = threadIdx.y;  // 32 x 8
#pragma unroll
  for (int j = 0; j < 32; j += 8)
    tile[ty + j][tx] = in[(size_t)(k0 + ty + j) * N + n0 + tx];
  __syncthreads();
#pragma unroll
  for (int j = 0; j < 32; j += 8)
    out[(size_t)(n0 + ty + j) * K + k0 + tx] = f32_to_bf16(tile[tx][ty + j]);
}

// ---------------------------------------------------------------------------
// Token + position embedding (fp32)
// ---------------------------------------------------------------------------
__global__ __launch_bounds__(256) void embed_kernel(
    const int* __restrict__ tokens, const float* __restrict__ wte,
    const float* __restrict__ wpe, float* __restrict__ x, int T, int E) {
  int row = blockIdx.x;            // b*T + t
  int t = row & (T - 1);
  int tok = tokens[row];
  const float* we = wte + (size_t)tok * E;
  const float* wp = wpe + (size_t)t * E;
  float* xr = x + (size_t)row * E;
  for (int e = threadIdx.x; e < E; e += blockDim.x) xr[e] = we[e] + wp[e];
}

// ---------------------------------------------------------------------------
// LayerNorm (fp32 in, bf16 out) — one block per token row
// ---------------------------------------------------------------------------
__global__ __launch_bounds__(256) void ln_kernel(
    const float* __restrict__ x, const float* __restrict__ g,
    const float* __restrict__ b, u16* __restrict__ out, int E) {
  int row = blockIdx.x;
  const float* xr = x + (size_t)row * E;
  __shared__ float red[256];
  float s = 0.f, s2 = 0.f;
  for (int e = threadIdx.x; e < E; e += 256) {
    float v = xr[e];
    s += v; s2 += v * v;
  }
  red[threadIdx.x] = s; __syncthreads();
  for (int o = 128; o > 0; o >>= 1) {
    if (threadIdx.x < o) red[threadIdx.x] += red[threadIdx.x + o];
    __syncthreads();
  }
  float mu = red[0] / (float)E;
  __syncthreads();
  red[threadIdx.x] = s2; __syncthreads();
  for (int o = 128; o > 0; o >>= 1) {
    if (threadIdx.x < o) red[threadIdx.x] += red[threadIdx.x + o];
    __syncthreads();
  }
  float var = red[0] / (float)E - mu * mu;
  float rs = rsqrtf(var + 1e-5f);
  u16* orow = out + (size_t)row * E;
  for (int e = threadIdx.x; e < 256 * ((E + 255) / 256); e += 256)
    if (e < E) orow[e] = f32_to_bf16((xr[e] - mu) * rs * g[e] + b[e]);
}

// ---------------------------------------------------------------------------
// Tiled bf16 GEMM with WMMA: C[M][N] = A[M][K] * Bt[N][K]^T (+bias/relu/resid)
// Block tile 128x256, 8 waves (2x4), wave tile 64x64 = 4x4 WMMA 16x16x32 tiles.
// Global->LDS via double-buffered GLOBAL_LOAD_ASYNC_TO_LDS_B128 (ASYNCcnt).
// ---------------------------------------------------------------------------
#define GBM 128
#define GBN 256
#define GBK 32

__global__ __launch_bounds__(256) void gemm_bf16_kernel(
    const u16* __restrict__ A, const u16* __restrict__ Bt,
    const float* __restrict__ bias, const float* __restrict__ resid,
    float* __restrict__ outF, u16* __restrict__ outH,
    int M, int N, int K, int flags) {
  __shared__ __align__(16) u16 lA[2][GBM * GBK];
  __shared__ __align__(16) u16 lB[2][GBN * GBK];
  int m0 = blockIdx.y * GBM, n0 = blockIdx.x * GBN;
  int tid = threadIdx.x;
  int lane = tid & 31, wave = tid >> 5;
  int waveM = wave >> 2;   // 0..1, 64 rows each
  int waveN = wave & 3;    // 0..3, 64 cols each

  v8f acc[4][4];
#pragma unroll
  for (int tr = 0; tr < 4; tr++)
#pragma unroll
    for (int tc = 0; tc < 4; tc++) acc[tr][tc] = (v8f)0.f;

  // Issue the 6 per-thread async b128 copies (2 for A, 4 for B) for K-step `s`.
  auto issue = [&](int s, int buf) {
    int k0 = s * GBK;
#pragma unroll
    for (int i = 0; i < 2; i++) {           // A: 128 rows x 4 chunks = 512
      int c = tid + i * 256;
      int r = c >> 2, cc = (c & 3) * 8;
      async_ld16((unsigned)(size_t)&lA[buf][r * GBK + cc],
                 &A[(size_t)(m0 + r) * K + k0 + cc]);
    }
#pragma unroll
    for (int i = 0; i < 4; i++) {           // B: 256 rows x 4 chunks = 1024
      int c = tid + i * 256;
      int r = c >> 2, cc = (c & 3) * 8;
      async_ld16((unsigned)(size_t)&lB[buf][r * GBK + cc],
                 &Bt[(size_t)(n0 + r) * K + k0 + cc]);
    }
  };

  int nk = K / GBK;
  issue(0, 0);
  for (int s = 0; s < nk; ++s) {
    int buf = s & 1;
    if (s + 1 < nk) {
      issue(s + 1, buf ^ 1);     // prefetch next tile into the other buffer
      WAIT_ASYNC(0x6);           // retire current tile's 6 async ops
    } else {
      WAIT_ASYNC(0x0);
    }
    __syncthreads();             // all waves' tiles visible in LDS

    v16bf aF[4], bF[4];
#pragma unroll
    for (int tr = 0; tr < 4; tr++)
      aF[tr] = ld_frag(lA[buf], GBK, waveM * 64 + tr * 16 + (lane & 15), 0, lane);
#pragma unroll
    for (int tc = 0; tc < 4; tc++)
      bF[tc] = ld_frag(lB[buf], GBK, waveN * 64 + tc * 16 + (lane & 15), 0, lane);
#pragma unroll
    for (int tr = 0; tr < 4; tr++)
#pragma unroll
      for (int tc = 0; tc < 4; tc++)
        acc[tr][tc] = WMMA_BF16(aF[tr], bF[tc], acc[tr][tc]);
    __syncthreads();             // done reading this buffer before it is refilled
  }

  // Epilogue. C layout: vgpr i, lane<16 -> row i, lane>=16 -> row 8+i; col = lane&15.
  int n = lane & 15, half = lane >> 4;
#pragma unroll
  for (int tr = 0; tr < 4; tr++) {
#pragma unroll
    for (int tc = 0; tc < 4; tc++) {
      int gcol = n0 + waveN * 64 + tc * 16 + n;
      float bv = (flags & FLAG_BIAS) ? bias[gcol] : 0.f;
      union { v8f v; float f[8]; } cu;
      cu.v = acc[tr][tc];
#pragma unroll
      for (int i = 0; i < 8; i++) {
        int grow = m0 + waveM * 64 + tr * 16 + half * 8 + i;
        size_t o = (size_t)grow * N + gcol;
        float v = cu.f[i] + bv;
        if (flags & FLAG_RELU) v = fmaxf(v, 0.f);
        if (flags & FLAG_RESID) v += resid[o];
        if (flags & FLAG_BF16) outH[o] = f32_to_bf16(v);
        else                   outF[o] = v;
      }
    }
  }
}

// ---------------------------------------------------------------------------
// Causal attention, flash-style: 64-query block per workgroup, 32-key chunks.
// qkv: bf16 [B*T][3E]; outO: bf16 [B*T][E]. S and softmax state in fp32 LDS.
// ---------------------------------------------------------------------------
__global__ __launch_bounds__(256) void attention_kernel(
    const u16* __restrict__ qkv, u16* __restrict__ outO) {
  const int T_ = 1024, E_ = 1024, DH_ = 64;
  int blk = blockIdx.x;
  int qb = blk & 15;          // 16 query blocks of 64
  int h  = (blk >> 4) & 15;   // 16 heads
  int b  = blk >> 8;          // 4 batches
  int qbase = qb * 64;

  const u16* Qp = qkv + (size_t)(b * T_) * 3 * E_ + h * DH_;
  const u16* Kp = Qp + E_;
  const u16* Vp = Qp + 2 * E_;

  __shared__ __align__(16) u16  lQ[64 * 64];
  __shared__ __align__(16) u16  lK[32 * 64];
  __shared__ __align__(16) u16  lVt[64 * 32];   // [d][key]
  __shared__ __align__(16) float lS[64 * 32];
  __shared__ __align__(16) u16  lP[64 * 32];
  __shared__ float rowM[64], rowSum[64], rowFac[64];

  int tid = threadIdx.x, lane = tid & 31, wave = tid >> 5;

  // Load Q tile 64x64 (row stride in global = 3E)
#pragma unroll
  for (int i = 0; i < 2; i++) {
    int c = tid + i * 256;            // 512 chunks of 8
    int r = c >> 3, cc = (c & 7) * 8;
    *reinterpret_cast<uint4*>(&lQ[r * 64 + cc]) =
        *reinterpret_cast<const uint4*>(&Qp[(size_t)(qbase + r) * 3 * E_ + cc]);
  }
  if (tid < 64) { rowM[tid] = -__builtin_inff(); rowSum[tid] = 0.f; }

  v8f oacc[2];
  oacc[0] = (v8f)0.f; oacc[1] = (v8f)0.f;
  int orow = (wave >> 1) * 16;    // wave's 16 output rows
  int ocol = (wave & 1) * 32;     // wave's 32 output cols (2 WMMA tiles)
  __syncthreads();

  const float scale = 0.125f;     // DH^-0.5
  for (int kb = 0; kb <= qbase + 63; kb += 32) {
    // K chunk 32x64
    {
      int r = tid >> 3, cc = (tid & 7) * 8;
      *reinterpret_cast<uint4*>(&lK[r * 64 + cc]) =
          *reinterpret_cast<const uint4*>(&Kp[(size_t)(kb + r) * 3 * E_ + cc]);
    }
    // V chunk transposed into lVt[d][key]
    {
      int key = tid >> 3;
      int d0 = (tid & 7) * 8;
      union { uint4 q; u16 hh[8]; } vt;
      vt.q = *reinterpret_cast<const uint4*>(&Vp[(size_t)(kb + key) * 3 * E_ + d0]);
#pragma unroll
      for (int j = 0; j < 8; j++) lVt[(d0 + j) * 32 + key] = vt.hh[j];
    }
    __syncthreads();

    // S = Q * K^T : 8 waves -> 8 tiles covering 64q x 32k
    {
      int qr = wave >> 1, kc = wave & 1;
      v8f sacc = (v8f)0.f;
#pragma unroll
      for (int ks = 0; ks < 64; ks += 32) {
        v16bf qf = ld_frag(lQ, 64, qr * 16 + (lane & 15), ks, lane);
        v16bf kf = ld_frag(lK, 64, kc * 16 + (lane & 15), ks, lane);
        sacc = WMMA_BF16(qf, kf, sacc);
      }
      int n = lane & 15, half = lane >> 4;
      union { v8f v; float f[8]; } su;
      su.v = sacc;
#pragma unroll
      for (int i = 0; i < 8; i++)
        lS[(qr * 16 + half * 8 + i) * 32 + kc * 16 + n] = su.f[i];
    }
    __syncthreads();

    // Online softmax row pass (64 rows handled by 64 threads)
    if (tid < 64) {
      int row = tid, qg = qbase + row;
      int lim = qg - kb + 1; if (lim > 32) lim = 32;
      float fac = 1.f;
      if (lim <= 0) {
#pragma unroll
        for (int j = 0; j < 32; j++) lP[row * 32 + j] = 0;
      } else {
        float m = rowM[row];
        float mx = -__builtin_inff();
        for (int j = 0; j < lim; j++) mx = fmaxf(mx, lS[row * 32 + j] * scale);
        float nm = fmaxf(m, mx);
        fac = expf(m - nm);
        float s = 0.f;
        for (int j = 0; j < 32; j++) {
          float p = (j < lim) ? expf(lS[row * 32 + j] * scale - nm) : 0.f;
          s += p;
          lP[row * 32 + j] = f32_to_bf16(p);
        }
        rowSum[row] = rowSum[row] * fac + s;
        rowM[row] = nm;
      }
      rowFac[row] = fac;
    }
    __syncthreads();

    // O = O*fac + P*V : each wave owns 16x32 of O (2 WMMA tiles)
    {
      int half = lane >> 4;
#pragma unroll
      for (int tc = 0; tc < 2; tc++) {
        union { v8f v; float f[8]; } ou;
        ou.v = oacc[tc];
#pragma unroll
        for (int i = 0; i < 8; i++) ou.f[i] *= rowFac[orow + half * 8 + i];
        v16bf pf = ld_frag(lP, 32, orow + (lane & 15), 0, lane);
        v16bf vf = ld_frag(lVt, 32, ocol + tc * 16 + (lane & 15), 0, lane);
        oacc[tc] = WMMA_BF16(pf, vf, ou.v);
      }
    }
    __syncthreads();
  }

  // Normalize and write O (bf16) at columns h*DH + ...
  {
    int n = lane & 15, half = lane >> 4;
#pragma unroll
    for (int tc = 0; tc < 2; tc++) {
      union { v8f v; float f[8]; } ou;
      ou.v = oacc[tc];
#pragma unroll
      for (int i = 0; i < 8; i++) {
        int row = orow + half * 8 + i;
        float val = ou.f[i] / rowSum[row];
        int col = h * DH_ + ocol + tc * 16 + n;
        outO[(size_t)(b * T_ + qbase + row) * E_ + col] = f32_to_bf16(val);
      }
    }
  }
}

// ---------------------------------------------------------------------------
// Host launcher
// ---------------------------------------------------------------------------
extern "C" void kernel_launch(void* const* d_in, const int* in_sizes, int n_in,
                              void* d_out, int out_size, void* d_ws, size_t ws_size,
                              hipStream_t stream) {
  (void)in_sizes; (void)n_in; (void)out_size; (void)ws_size;
  const int Bsz = 4, T = 1024, E = 1024, L = 8, V = 32000;
  const int BT = Bsz * T;
  const int E3 = 3 * E, E4 = 4 * E;

  const int*   tokens = (const int*)  d_in[0];
  const float* wte    = (const float*)d_in[1];
  const float* wpe    = (const float*)d_in[2];
  const float* qkv_w  = (const float*)d_in[3];
  const float* proj_w = (const float*)d_in[4];
  const float* proj_b = (const float*)d_in[5];
  const float* ff1_w  = (const float*)d_in[6];
  const float* ff1_b  = (const float*)d_in[7];
  const float* ff2_w  = (const float*)d_in[8];
  const float* ff2_b  = (const float*)d_in[9];
  const float* ln1_g  = (const float*)d_in[10];
  const float* ln1_b  = (const float*)d_in[11];
  const float* ln2_g  = (const float*)d_in[12];
  const float* ln2_b  = (const float*)d_in[13];
  const float* lnf_g  = (const float*)d_in[14];
  const float* lnf_b  = (const float*)d_in[15];
  const float* head_w = (const float*)d_in[16];
  const float* head_b = (const float*)d_in[17];
  float* logits = (float*)d_out;

  char* ws = (char*)d_ws;
  size_t off = 0;
  auto take = [&](size_t bytes) { char* p = ws + off; off += (bytes + 255) & ~(size_t)255; return p; };
  u16* qkvT  = (u16*)take((size_t)L * E3 * E * 2);   // [l][3E][E]
  u16* projT = (u16*)take((size_t)L * E * E * 2);    // [l][E][E]
  u16* ff1T  = (u16*)take((size_t)L * E4 * E * 2);   // [l][4E][E]
  u16* ff2T  = (u16*)take((size_t)L * E * E4 * 2);   // [l][E][4E]
  u16* headT = (u16*)take((size_t)V * E * 2);        // [V][E]
  float* x   = (float*)take((size_t)BT * E * 4);
  u16* xn    = (u16*)take((size_t)BT * E * 2);
  u16* qkv   = (u16*)take((size_t)BT * E3 * 2);
  u16* attnO = (u16*)take((size_t)BT * E * 2);
  u16* hid   = (u16*)take((size_t)BT * E4 * 2);

  dim3 tb(32, 8);
  // Weight convert + transpose to Bt layout [N][K] bf16
  for (int l = 0; l < L; l++) {
    transpose_convert_kernel<<<dim3(E3 / 32, E / 32), tb, 0, stream>>>(
        qkv_w + (size_t)l * E * E3, qkvT + (size_t)l * E3 * E, E, E3);
    transpose_convert_kernel<<<dim3(E / 32, E / 32), tb, 0, stream>>>(
        proj_w + (size_t)l * E * E, projT + (size_t)l * E * E, E, E);
    transpose_convert_kernel<<<dim3(E4 / 32, E / 32), tb, 0, stream>>>(
        ff1_w + (size_t)l * E * E4, ff1T + (size_t)l * E4 * E, E, E4);
    transpose_convert_kernel<<<dim3(E / 32, E4 / 32), tb, 0, stream>>>(
        ff2_w + (size_t)l * E4 * E, ff2T + (size_t)l * E * E4, E4, E);
  }
  transpose_convert_kernel<<<dim3(V / 32, E / 32), tb, 0, stream>>>(head_w, headT, E, V);

  embed_kernel<<<BT, 256, 0, stream>>>(tokens, wte, wpe, x, T, E);

  for (int l = 0; l < L; l++) {
    ln_kernel<<<BT, 256, 0, stream>>>(x, ln1_g + (size_t)l * E, ln1_b + (size_t)l * E, xn, E);
    // qkv = xn @ qkv_w   (no bias), bf16 out
    gemm_bf16_kernel<<<dim3(E3 / GBN, BT / GBM), 256, 0, stream>>>(
        xn, qkvT + (size_t)l * E3 * E, nullptr, nullptr, nullptr, qkv,
        BT, E3, E, FLAG_BF16);
    attention_kernel<<<Bsz * 16 * (T / 64), 256, 0, stream>>>(qkv, attnO);
    // x = x + attnO @ proj_w + proj_b   (fp32 residual)
    gemm_bf16_kernel<<<dim3(E / GBN, BT / GBM), 256, 0, stream>>>(
        attnO, projT + (size_t)l * E * E, proj_b + (size_t)l * E, x, x, nullptr,
        BT, E, E, FLAG_RESID | FLAG_BIAS);
    ln_kernel<<<BT, 256, 0, stream>>>(x, ln2_g + (size_t)l * E, ln2_b + (size_t)l * E, xn, E);
    // hid = relu(xn @ ff1_w + ff1_b), bf16 out
    gemm_bf16_kernel<<<dim3(E4 / GBN, BT / GBM), 256, 0, stream>>>(
        xn, ff1T + (size_t)l * E4 * E, ff1_b + (size_t)l * E4, nullptr, nullptr, hid,
        BT, E4, E, FLAG_RELU | FLAG_BIAS | FLAG_BF16);
    // x = x + hid @ ff2_w + ff2_b
    gemm_bf16_kernel<<<dim3(E / GBN, BT / GBM), 256, 0, stream>>>(
        hid, ff2T + (size_t)l * E * E4, ff2_b + (size_t)l * E, x, x, nullptr,
        BT, E, E4, FLAG_RESID | FLAG_BIAS);
  }

  ln_kernel<<<BT, 256, 0, stream>>>(x, lnf_g, lnf_b, xn, E);
  // logits = xn @ head_w + head_b
  gemm_bf16_kernel<<<dim3(V / GBN, BT / GBM), 256, 0, stream>>>(
      xn, headT, head_b, nullptr, logits, nullptr,
      BT, V, E, FLAG_BIAS);
}
